// Llama4MoE_83580063580300
// MI455X (gfx1250) — matrix operand order repaired and checked by reading, not compiled, not run on hardware
//
#include <hip/hip_runtime.h>

#define T_TOK 4096
#define H_DIM 2048
#define E_NUM 16
#define I_DIM 1024
#define SI_DIM 4096

#define BM 128
#define BN 64
#define BK 32
#define BKP (BK + 8)   // LDS pad: 80B rows, keeps 8/16B alignment for b64/b128 ops

typedef __attribute__((ext_vector_type(16))) __bf16 v16bf;
typedef __attribute__((ext_vector_type(4)))  __bf16 v4bf;
typedef __attribute__((ext_vector_type(8)))  float  v8f;

__device__ __forceinline__ v8f wmma_bf16(v16bf a, v16bf b, v8f c) {
  // 8 args: (neg_a, A, neg_b, B, c_mod, C, reuse_a, reuse_b)
  return __builtin_amdgcn_wmma_f32_16x16x32_bf16(false, a, false, b, (short)0, c,
                                                 false, false);
}

__device__ __forceinline__ v8f zero_v8f() {
  v8f z;
#pragma unroll
  for (int i = 0; i < 8; ++i) z[i] = 0.0f;
  return z;
}

__device__ __forceinline__ float silu_f(float g) {
  return g * (1.0f / (1.0f + __expf(-g)));
}

__device__ __forceinline__ v4bf pack4(float4 v) {
  v4bf p;
  p[0] = (__bf16)v.x; p[1] = (__bf16)v.y; p[2] = (__bf16)v.z; p[3] = (__bf16)v.w;
  return p;
}

// A tile: 4 K-contiguous values of one row -> single b64 LDS store
__device__ __forceinline__ void st_a4(__bf16* As, int r, int c, float4 v) {
  *(v4bf*)&As[r * BKP + c] = pack4(v);
}

// B micro-block: thread holds 4 K-rows x 4 N-cols; emit one b64 store per column
// (K-contiguous in the N-major tile) -> fully packed cvt_pk + ds_store_b64.
__device__ __forceinline__ void st_b16x(__bf16* Bs, int nq, int kq,
                                        float4 r0, float4 r1, float4 r2, float4 r3) {
  const float* q0 = (const float*)&r0;
  const float* q1 = (const float*)&r1;
  const float* q2 = (const float*)&r2;
  const float* q3 = (const float*)&r3;
#pragma unroll
  for (int nn = 0; nn < 4; ++nn) {
    v4bf t;
    t[0] = (__bf16)q0[nn]; t[1] = (__bf16)q1[nn];
    t[2] = (__bf16)q2[nn]; t[3] = (__bf16)q3[nn];
    *(v4bf*)&Bs[(nq + nn) * BKP + kq] = t;
  }
}

// ---- fragment builders per CDNA5 WMMA VGPR layouts (wave32) ----

// A 16x32 bf16: lanes 0-15 row M=lane, K = {0..7,16..23}; lanes 16-31 row M=lane-16,
// K = {8..15,24..31}. LDS A tile is row-major [BM][BKP].
__device__ __forceinline__ v16bf load_a_frag(const __bf16* base, int row0, int lane) {
  const __bf16* p = base + (row0 + (lane & 15)) * BKP;
  const int kb = (lane >> 4) * 8;
  v16bf a;
#pragma unroll
  for (int i = 0; i < 8; ++i) { a[i] = p[kb + i]; a[i + 8] = p[16 + kb + i]; }
  return a;
}

// B 32x16 bf16: lanes 0-15 col N=lane, K = {0..7,16..23}; lanes 16-31 col N=lane-16,
// K = {8..15,24..31}. LDS B tile stored N-major [BN][BKP].
__device__ __forceinline__ v16bf load_b_frag(const __bf16* base, int col0, int lane) {
  const __bf16* p = base + (col0 + (lane & 15)) * BKP;
  const int kb = (lane >> 4) * 8;
  v16bf b;
#pragma unroll
  for (int i = 0; i < 8; ++i) { b[i] = p[kb + i]; b[i + 8] = p[16 + kb + i]; }
  return b;
}

// ======================= router (top-1 + sigmoid) =======================
__global__ __launch_bounds__(256) void k_router(const float* __restrict__ x,
                                                const float* __restrict__ rw,
                                                int* __restrict__ eid,
                                                float* __restrict__ wgt) {
  const int t = blockIdx.x;
  const int tid = threadIdx.x;
  const int e = tid & 15;   // expert
  const int s = tid >> 4;   // slice
  const float* xr = x + (size_t)t * H_DIM;
  const float* wr = rw + (size_t)e * H_DIM;
  float acc = 0.0f;
  for (int k = s; k < H_DIM; k += 16) acc += xr[k] * wr[k];
  __shared__ float part[256];
  part[tid] = acc;
  __syncthreads();
  if (tid < 16) {
    float v = 0.0f;
#pragma unroll
    for (int j = 0; j < 16; ++j) v += part[tid + j * 16];
    part[tid] = v;
  }
  __syncthreads();
  if (tid == 0) {
    int be = 0;
    float bv = part[0];
#pragma unroll
    for (int j = 1; j < 16; ++j) if (part[j] > bv) { bv = part[j]; be = j; }
    eid[t] = be;
    wgt[t] = 1.0f / (1.0f + __expf(-bv));
  }
}

__global__ void k_zero_cnt(int* __restrict__ cnt) {
  if (threadIdx.x < E_NUM) cnt[threadIdx.x] = 0;
}

__global__ void k_assign(const int* __restrict__ eid, int* __restrict__ cnt,
                         int* __restrict__ list) {
  const int t = blockIdx.x * blockDim.x + threadIdx.x;
  if (t < T_TOK) {
    const int e = eid[t];
    const int s = atomicAdd(&cnt[e], 1);
    list[e * T_TOK + s] = t;
  }
}

__global__ void k_prefix(const int* __restrict__ cnt, int* __restrict__ toff) {
  if (threadIdx.x == 0) {
    int acc = 0;
    for (int e = 0; e < E_NUM; ++e) { toff[e] = acc; acc += (cnt[e] + BM - 1) / BM; }
    toff[E_NUM] = acc;
  }
}

// =============== shared expert: fused gate/up + SiLU -> bf16 act_s ===============
__global__ __launch_bounds__(256) void k_shared_gateup(const float* __restrict__ x,
                                                       const float* __restrict__ wsg,
                                                       const float* __restrict__ wsu,
                                                       __bf16* __restrict__ act_s) {
  __shared__ __align__(16) __bf16 As[2][BM * BKP];
  __shared__ __align__(16) __bf16 Bg[2][BN * BKP];
  __shared__ __align__(16) __bf16 Bu[2][BN * BKP];
  const int tid = threadIdx.x, lane = tid & 31, wave = tid >> 5;
  const int mt = blockIdx.x & 31;   // T/BM = 32
  const int nt = blockIdx.x >> 5;   // SI/BN = 64
  const int row0 = mt * BM, col0 = nt * BN;
  const int rb = (wave >> 1) * 32, cb = (wave & 1) * 32;

  // A loader coords: 4x float4 per thread, K-contiguous
  int ar[4], ac[4];
#pragma unroll
  for (int i = 0; i < 4; ++i) { const int idx = tid + i * 256; ar[i] = idx >> 3; ac[i] = (idx & 7) * 4; }
  // B loader: lower half -> gate, upper half -> up; 4x4 micro-block per thread
  const int bj = tid & 127;
  const int bnq = (bj & 15) * 4;   // local col base (16 groups * 4 = 64)
  const int bkq = (bj >> 4) * 4;   // local k base   (8 groups * 4 = 32)
  const float* bsrc = (tid < 128) ? wsg : wsu;

  float4 pa[4], pb[4];
  // prologue: stage K tile 0
#pragma unroll
  for (int i = 0; i < 4; ++i)
    pa[i] = *(const float4*)&x[(size_t)(row0 + ar[i]) * H_DIM + ac[i]];
#pragma unroll
  for (int q = 0; q < 4; ++q)
    pb[q] = *(const float4*)&bsrc[(size_t)(bkq + q) * SI_DIM + col0 + bnq];
#pragma unroll
  for (int i = 0; i < 4; ++i) st_a4(As[0], ar[i], ac[i], pa[i]);
  st_b16x((tid < 128) ? Bg[0] : Bu[0], bnq, bkq, pb[0], pb[1], pb[2], pb[3]);
  __syncthreads();

  v8f cg[2][2], cu[2][2];
#pragma unroll
  for (int i = 0; i < 2; ++i)
#pragma unroll
    for (int j = 0; j < 2; ++j) { cg[i][j] = zero_v8f(); cu[i][j] = zero_v8f(); }

  int buf = 0;
  for (int k0 = 0; k0 < H_DIM; k0 += BK) {
    const bool more = (k0 + BK) < H_DIM;
    if (more) {  // prefetch tile k+1 into registers (overlaps with WMMA below)
      const int kn = k0 + BK;
#pragma unroll
      for (int i = 0; i < 4; ++i)
        pa[i] = *(const float4*)&x[(size_t)(row0 + ar[i]) * H_DIM + kn + ac[i]];
#pragma unroll
      for (int q = 0; q < 4; ++q)
        pb[q] = *(const float4*)&bsrc[(size_t)(kn + bkq + q) * SI_DIM + col0 + bnq];
    }
    const v16bf a0 = load_a_frag(As[buf], rb, lane);
    const v16bf a1 = load_a_frag(As[buf], rb + 16, lane);
    const v16bf g0 = load_b_frag(Bg[buf], cb, lane);
    const v16bf g1 = load_b_frag(Bg[buf], cb + 16, lane);
    const v16bf u0 = load_b_frag(Bu[buf], cb, lane);
    const v16bf u1 = load_b_frag(Bu[buf], cb + 16, lane);
    cg[0][0] = wmma_bf16(a0, g0, cg[0][0]);
    cg[0][1] = wmma_bf16(a0, g1, cg[0][1]);
    cg[1][0] = wmma_bf16(a1, g0, cg[1][0]);
    cg[1][1] = wmma_bf16(a1, g1, cg[1][1]);
    cu[0][0] = wmma_bf16(a0, u0, cu[0][0]);
    cu[0][1] = wmma_bf16(a0, u1, cu[0][1]);
    cu[1][0] = wmma_bf16(a1, u0, cu[1][0]);
    cu[1][1] = wmma_bf16(a1, u1, cu[1][1]);
    if (more) {  // drain prefetch into the other buffer
#pragma unroll
      for (int i = 0; i < 4; ++i) st_a4(As[buf ^ 1], ar[i], ac[i], pa[i]);
      st_b16x((tid < 128) ? Bg[buf ^ 1] : Bu[buf ^ 1], bnq, bkq, pb[0], pb[1], pb[2], pb[3]);
    }
    __syncthreads();   // single barrier per K step (ping-pong buffers)
    buf ^= 1;
  }

#pragma unroll
  for (int mf = 0; mf < 2; ++mf)
#pragma unroll
    for (int nf = 0; nf < 2; ++nf)
#pragma unroll
      for (int r = 0; r < 8; ++r) {
        const int row = row0 + rb + mf * 16 + ((lane >> 4) * 8) + r;
        const int col = col0 + cb + nf * 16 + (lane & 15);
        const float a = silu_f(cg[mf][nf][r]) * cu[mf][nf][r];
        act_s[(size_t)row * SI_DIM + col] = (__bf16)a;
      }
}

// =============== shared expert: down proj, writes d_out ===============
__global__ __launch_bounds__(256) void k_shared_down(const __bf16* __restrict__ act_s,
                                                     const float* __restrict__ wsd,
                                                     float* __restrict__ out) {
  __shared__ __align__(16) __bf16 As[2][BM * BKP];
  __shared__ __align__(16) __bf16 Bs[2][BN * BKP];
  const int tid = threadIdx.x, lane = tid & 31, wave = tid >> 5;
  const int mt = blockIdx.x & 31;   // T/BM = 32
  const int nt = blockIdx.x >> 5;   // H/BN = 32
  const int row0 = mt * BM, col0 = nt * BN;
  const int rb = (wave >> 1) * 32, cb = (wave & 1) * 32;

  int ar[2], ac8[2];
#pragma unroll
  for (int i = 0; i < 2; ++i) {
    const int idx = tid + i * 256;
    ar[i] = idx >> 2; ac8[i] = (idx & 3) * 8;      // A: 128x32 bf16 as uint4
  }
  const int bnq = (tid & 15) * 4;                  // B by lower 128 threads, 4x4 blocks
  const int bkq = ((tid & 127) >> 4) * 4;
  const bool bown = tid < 128;

  uint4 pa[2];
  float4 pb[4];
#pragma unroll
  for (int i = 0; i < 2; ++i)
    pa[i] = *(const uint4*)&act_s[(size_t)(row0 + ar[i]) * SI_DIM + ac8[i]];
  if (bown) {
#pragma unroll
    for (int q = 0; q < 4; ++q)
      pb[q] = *(const float4*)&wsd[(size_t)(bkq + q) * H_DIM + col0 + bnq];
  }
#pragma unroll
  for (int i = 0; i < 2; ++i) *(uint4*)&As[0][ar[i] * BKP + ac8[i]] = pa[i];
  if (bown) st_b16x(Bs[0], bnq, bkq, pb[0], pb[1], pb[2], pb[3]);
  __syncthreads();

  v8f c[2][2];
#pragma unroll
  for (int i = 0; i < 2; ++i)
#pragma unroll
    for (int j = 0; j < 2; ++j) c[i][j] = zero_v8f();

  int buf = 0;
  for (int k0 = 0; k0 < SI_DIM; k0 += BK) {
    const bool more = (k0 + BK) < SI_DIM;
    if (more) {
      const int kn = k0 + BK;
#pragma unroll
      for (int i = 0; i < 2; ++i)
        pa[i] = *(const uint4*)&act_s[(size_t)(row0 + ar[i]) * SI_DIM + kn + ac8[i]];
      if (bown) {
#pragma unroll
        for (int q = 0; q < 4; ++q)
          pb[q] = *(const float4*)&wsd[(size_t)(kn + bkq + q) * H_DIM + col0 + bnq];
      }
    }
    const v16bf a0 = load_a_frag(As[buf], rb, lane);
    const v16bf a1 = load_a_frag(As[buf], rb + 16, lane);
    const v16bf b0 = load_b_frag(Bs[buf], cb, lane);
    const v16bf b1 = load_b_frag(Bs[buf], cb + 16, lane);
    c[0][0] = wmma_bf16(a0, b0, c[0][0]);
    c[0][1] = wmma_bf16(a0, b1, c[0][1]);
    c[1][0] = wmma_bf16(a1, b0, c[1][0]);
    c[1][1] = wmma_bf16(a1, b1, c[1][1]);
    if (more) {
#pragma unroll
      for (int i = 0; i < 2; ++i) *(uint4*)&As[buf ^ 1][ar[i] * BKP + ac8[i]] = pa[i];
      if (bown) st_b16x(Bs[buf ^ 1], bnq, bkq, pb[0], pb[1], pb[2], pb[3]);
    }
    __syncthreads();
    buf ^= 1;
  }

#pragma unroll
  for (int mf = 0; mf < 2; ++mf)
#pragma unroll
    for (int nf = 0; nf < 2; ++nf)
#pragma unroll
      for (int r = 0; r < 8; ++r) {
        const int row = row0 + rb + mf * 16 + ((lane >> 4) * 8) + r;
        const int col = col0 + cb + nf * 16 + (lane & 15);
        out[(size_t)row * H_DIM + col] = c[mf][nf][r];
      }
}

// ====== routed experts: grouped gate/up (gather + scale + SiLU) -> bf16 act_r ======
__global__ __launch_bounds__(256) void k_routed_gateup(
    const float* __restrict__ x, const float* __restrict__ wg,
    const float* __restrict__ wu, const int* __restrict__ list,
    const int* __restrict__ cnt, const int* __restrict__ toff,
    const float* __restrict__ wgt, __bf16* __restrict__ act_r) {
  __shared__ __align__(16) __bf16 As[2][BM * BKP];
  __shared__ __align__(16) __bf16 Bg[2][BN * BKP];
  __shared__ __align__(16) __bf16 Bu[2][BN * BKP];
  __shared__ int toks[BM];
  __shared__ float ww[BM];
  const int tid = threadIdx.x, lane = tid & 31, wave = tid >> 5;
  const int rb = (wave >> 1) * 32, cb = (wave & 1) * 32;
  const int NT = I_DIM / BN;  // 16
  const int totalM = toff[E_NUM];

  int ar[4], ac[4];
#pragma unroll
  for (int i = 0; i < 4; ++i) { const int idx = tid + i * 256; ar[i] = idx >> 3; ac[i] = (idx & 7) * 4; }
  const int bj = tid & 127;
  const int bnq = (bj & 15) * 4;
  const int bkq = (bj >> 4) * 4;

  for (int tile = blockIdx.x; tile < totalM * NT; tile += gridDim.x) {
    const int nt = tile % NT;
    const int mt = tile / NT;
    int e = 0;
#pragma unroll
    for (int j = 1; j < E_NUM; ++j) if (toff[j] <= mt) e = j;
    const int ml = mt - toff[e];
    const int ce = cnt[e];
    const int col0 = nt * BN;
    if (tid < BM) {
      const int gi = ml * BM + tid;
      const int tok = (gi < ce) ? list[e * T_TOK + gi] : -1;
      toks[tid] = tok;
      ww[tid] = (tok >= 0) ? wgt[tok] : 0.0f;
    }
    __syncthreads();   // toks visible to all loaders

    const float* bsrc = ((tid < 128) ? wg : wu) + (size_t)e * H_DIM * I_DIM;

    float4 pa[4], pb[4];
#pragma unroll
    for (int i = 0; i < 4; ++i) {
      const int tok = toks[ar[i]];
      pa[i] = make_float4(0.f, 0.f, 0.f, 0.f);
      if (tok >= 0) pa[i] = *(const float4*)&x[(size_t)tok * H_DIM + ac[i]];
    }
#pragma unroll
    for (int q = 0; q < 4; ++q)
      pb[q] = *(const float4*)&bsrc[(size_t)(bkq + q) * I_DIM + col0 + bnq];
#pragma unroll
    for (int i = 0; i < 4; ++i) st_a4(As[0], ar[i], ac[i], pa[i]);
    st_b16x((tid < 128) ? Bg[0] : Bu[0], bnq, bkq, pb[0], pb[1], pb[2], pb[3]);
    __syncthreads();

    v8f cg[2][2], cu[2][2];
#pragma unroll
    for (int i = 0; i < 2; ++i)
#pragma unroll
      for (int j = 0; j < 2; ++j) { cg[i][j] = zero_v8f(); cu[i][j] = zero_v8f(); }

    int buf = 0;
    for (int k0 = 0; k0 < H_DIM; k0 += BK) {
      const bool more = (k0 + BK) < H_DIM;
      if (more) {
        const int kn = k0 + BK;
#pragma unroll
        for (int i = 0; i < 4; ++i) {
          const int tok = toks[ar[i]];
          pa[i] = make_float4(0.f, 0.f, 0.f, 0.f);
          if (tok >= 0) pa[i] = *(const float4*)&x[(size_t)tok * H_DIM + kn + ac[i]];
        }
#pragma unroll
        for (int q = 0; q < 4; ++q)
          pb[q] = *(const float4*)&bsrc[(size_t)(kn + bkq + q) * I_DIM + col0 + bnq];
      }
      const v16bf a0 = load_a_frag(As[buf], rb, lane);
      const v16bf a1 = load_a_frag(As[buf], rb + 16, lane);
      const v16bf g0 = load_b_frag(Bg[buf], cb, lane);
      const v16bf g1 = load_b_frag(Bg[buf], cb + 16, lane);
      const v16bf u0 = load_b_frag(Bu[buf], cb, lane);
      const v16bf u1 = load_b_frag(Bu[buf], cb + 16, lane);
      cg[0][0] = wmma_bf16(a0, g0, cg[0][0]);
      cg[0][1] = wmma_bf16(a0, g1, cg[0][1]);
      cg[1][0] = wmma_bf16(a1, g0, cg[1][0]);
      cg[1][1] = wmma_bf16(a1, g1, cg[1][1]);
      cu[0][0] = wmma_bf16(a0, u0, cu[0][0]);
      cu[0][1] = wmma_bf16(a0, u1, cu[0][1]);
      cu[1][0] = wmma_bf16(a1, u0, cu[1][0]);
      cu[1][1] = wmma_bf16(a1, u1, cu[1][1]);
      if (more) {
#pragma unroll
        for (int i = 0; i < 4; ++i) st_a4(As[buf ^ 1], ar[i], ac[i], pa[i]);
        st_b16x((tid < 128) ? Bg[buf ^ 1] : Bu[buf ^ 1], bnq, bkq, pb[0], pb[1], pb[2], pb[3]);
      }
      __syncthreads();
      buf ^= 1;
    }

#pragma unroll
    for (int mf = 0; mf < 2; ++mf)
#pragma unroll
      for (int nf = 0; nf < 2; ++nf)
#pragma unroll
        for (int r = 0; r < 8; ++r) {
          const int lr = rb + mf * 16 + ((lane >> 4) * 8) + r;
          const int tok = toks[lr];
          if (tok >= 0) {
            const float d = ww[lr];
            const float g = cg[mf][nf][r] * d;   // scale BEFORE SiLU (matches ref)
            const float u = cu[mf][nf][r] * d;
            const int col = col0 + cb + nf * 16 + (lane & 15);
            act_r[(size_t)tok * I_DIM + col] = (__bf16)(silu_f(g) * u);
          }
        }
    __syncthreads();   // protect toks/ww/LDS before next persistent tile
  }
}

// ====== routed experts: grouped down proj, scatter-add into d_out ======
__global__ __launch_bounds__(256) void k_routed_down(
    const __bf16* __restrict__ act_r, const float* __restrict__ wd,
    const int* __restrict__ list, const int* __restrict__ cnt,
    const int* __restrict__ toff, float* __restrict__ out) {
  __shared__ __align__(16) __bf16 As[2][BM * BKP];
  __shared__ __align__(16) __bf16 Bs[2][BN * BKP];
  __shared__ int toks[BM];
  const int tid = threadIdx.x, lane = tid & 31, wave = tid >> 5;
  const int rb = (wave >> 1) * 32, cb = (wave & 1) * 32;
  const int NT = H_DIM / BN;  // 32
  const int totalM = toff[E_NUM];

  int ar[2], ac8[2];
#pragma unroll
  for (int i = 0; i < 2; ++i) {
    const int idx = tid + i * 256;
    ar[i] = idx >> 2; ac8[i] = (idx & 3) * 8;
  }
  const int bnq = (tid & 15) * 4;
  const int bkq = ((tid & 127) >> 4) * 4;
  const bool bown = tid < 128;

  for (int tile = blockIdx.x; tile < totalM * NT; tile += gridDim.x) {
    const int nt = tile % NT;
    const int mt = tile / NT;
    int e = 0;
#pragma unroll
    for (int j = 1; j < E_NUM; ++j) if (toff[j] <= mt) e = j;
    const int ml = mt - toff[e];
    const int ce = cnt[e];
    const int col0 = nt * BN;
    if (tid < BM) {
      const int gi = ml * BM + tid;
      toks[tid] = (gi < ce) ? list[e * T_TOK + gi] : -1;
    }
    __syncthreads();

    const float* wdb = wd + (size_t)e * I_DIM * H_DIM;

    uint4 pa[2];
    float4 pb[4];
#pragma unroll
    for (int i = 0; i < 2; ++i) {
      const int tok = toks[ar[i]];
      pa[i] = make_uint4(0u, 0u, 0u, 0u);
      if (tok >= 0) pa[i] = *(const uint4*)&act_r[(size_t)tok * I_DIM + ac8[i]];
    }
    if (bown) {
#pragma unroll
      for (int q = 0; q < 4; ++q)
        pb[q] = *(const float4*)&wdb[(size_t)(bkq + q) * H_DIM + col0 + bnq];
    }
#pragma unroll
    for (int i = 0; i < 2; ++i) *(uint4*)&As[0][ar[i] * BKP + ac8[i]] = pa[i];
    if (bown) st_b16x(Bs[0], bnq, bkq, pb[0], pb[1], pb[2], pb[3]);
    __syncthreads();

    v8f c[2][2];
#pragma unroll
    for (int i = 0; i < 2; ++i)
#pragma unroll
      for (int j = 0; j < 2; ++j) c[i][j] = zero_v8f();

    int buf = 0;
    for (int k0 = 0; k0 < I_DIM; k0 += BK) {
      const bool more = (k0 + BK) < I_DIM;
      if (more) {
        const int kn = k0 + BK;
#pragma unroll
        for (int i = 0; i < 2; ++i) {
          const int tok = toks[ar[i]];
          pa[i] = make_uint4(0u, 0u, 0u, 0u);
          if (tok >= 0) pa[i] = *(const uint4*)&act_r[(size_t)tok * I_DIM + kn + ac8[i]];
        }
        if (bown) {
#pragma unroll
          for (int q = 0; q < 4; ++q)
            pb[q] = *(const float4*)&wdb[(size_t)(kn + bkq + q) * H_DIM + col0 + bnq];
        }
      }
      const v16bf a0 = load_a_frag(As[buf], rb, lane);
      const v16bf a1 = load_a_frag(As[buf], rb + 16, lane);
      const v16bf b0 = load_b_frag(Bs[buf], cb, lane);
      const v16bf b1 = load_b_frag(Bs[buf], cb + 16, lane);
      c[0][0] = wmma_bf16(a0, b0, c[0][0]);
      c[0][1] = wmma_bf16(a0, b1, c[0][1]);
      c[1][0] = wmma_bf16(a1, b0, c[1][0]);
      c[1][1] = wmma_bf16(a1, b1, c[1][1]);
      if (more) {
#pragma unroll
        for (int i = 0; i < 2; ++i) *(uint4*)&As[buf ^ 1][ar[i] * BKP + ac8[i]] = pa[i];
        if (bown) st_b16x(Bs[buf ^ 1], bnq, bkq, pb[0], pb[1], pb[2], pb[3]);
      }
      __syncthreads();
      buf ^= 1;
    }

#pragma unroll
    for (int mf = 0; mf < 2; ++mf)
#pragma unroll
      for (int nf = 0; nf < 2; ++nf)
#pragma unroll
        for (int r = 0; r < 8; ++r) {
          const int lr = rb + mf * 16 + ((lane >> 4) * 8) + r;
          const int tok = toks[lr];
          if (tok >= 0) {
            const int col = col0 + cb + nf * 16 + (lane & 15);
            // top-1: each (tok,col) owned by exactly one tile -> plain RMW is safe
            out[(size_t)tok * H_DIM + col] += c[mf][nf][r];
          }
        }
    __syncthreads();
  }
}

extern "C" void kernel_launch(void* const* d_in, const int* in_sizes, int n_in,
                              void* d_out, int out_size, void* d_ws, size_t ws_size,
                              hipStream_t stream) {
  const float* x   = (const float*)d_in[0];  // [T,H]
  const float* rw  = (const float*)d_in[1];  // [E,H]
  const float* wg  = (const float*)d_in[2];  // [E,H,I]
  const float* wu  = (const float*)d_in[3];  // [E,H,I]
  const float* wd  = (const float*)d_in[4];  // [E,I,H]
  const float* wsg = (const float*)d_in[5];  // [H,SI]
  const float* wsu = (const float*)d_in[6];  // [H,SI]
  const float* wsd = (const float*)d_in[7];  // [SI,H]
  float* out = (float*)d_out;                // [T,H]

  char* ws = (char*)d_ws;
  int*    eid   = (int*)(ws + 0);            // T ints
  float*  wgt   = (float*)(ws + 16384);      // T floats
  int*    cnt   = (int*)(ws + 32768);        // E ints
  int*    toff  = (int*)(ws + 33024);        // E+1 ints
  int*    list  = (int*)(ws + 33280);        // E*T ints (256 KB)
  __bf16* act_s = (__bf16*)(ws + 295424);    // T*SI bf16 (32 MB)
  __bf16* act_r = (__bf16*)(ws + 33849856);  // T*I  bf16 (8 MB)

  k_zero_cnt<<<1, 32, 0, stream>>>(cnt);
  k_router<<<T_TOK, 256, 0, stream>>>(x, rw, eid, wgt);
  k_assign<<<T_TOK / 256, 256, 0, stream>>>(eid, cnt, list);
  k_prefix<<<1, 32, 0, stream>>>(cnt, toff);
  k_shared_gateup<<<(T_TOK / BM) * (SI_DIM / BN), 256, 0, stream>>>(x, wsg, wsu, act_s);
  k_shared_down<<<(T_TOK / BM) * (H_DIM / BN), 256, 0, stream>>>(act_s, wsd, out);
  k_routed_gateup<<<512, 256, 0, stream>>>(x, wg, wu, list, cnt, toff, wgt, act_r);
  k_routed_down<<<512, 256, 0, stream>>>(act_r, wd, list, cnt, toff, out);
}
